// RNNHierEncoder_64390149701907
// MI455X (gfx1250) — matrix operand order repaired and checked by reading, not compiled, not run on hardware
//
#include <hip/hip_runtime.h>

// Problem constants (from reference): S=512, N=256, E=1024
#define S_LEN 512
#define N_B   256
#define E_DIM 1024
#define G_DIM 4096   // 4*E

typedef __attribute__((ext_vector_type(16))) __bf16          v16bf;
typedef __attribute__((ext_vector_type(16))) unsigned short  v16us;
typedef __attribute__((ext_vector_type(8)))  unsigned short  v8us;
typedef __attribute__((ext_vector_type(4)))  unsigned short  v4us;
typedef __attribute__((ext_vector_type(8)))  float           v8f;

// fp32 -> bf16 bits, round-to-nearest-even
__device__ __forceinline__ unsigned short f2bf(float x) {
  unsigned int u = __builtin_bit_cast(unsigned int, x);
  u += 0x7FFFu + ((u >> 16) & 1u);
  return (unsigned short)(u >> 16);
}

__device__ __forceinline__ float sigm(float x) {
  return 1.0f / (1.0f + __expf(-x));
}

// ---- WMMA fragment loaders (all operands bf16 in memory) -------------------
// A-matrix 16x32 bf16 layout (ISA 7.12.2): lane holds row M = lane&15;
// lane-half lh: element e (0..15): K = k0 + (e>=8 ? 16 : 0) + lh*8 + (e&7)
// -> two contiguous 8-element (16B) runs per lane.
__device__ __forceinline__ v16bf load_a_frag(const unsigned short* __restrict__ rowp,
                                             int k0, int lh) {
  const unsigned short* q = rowp + k0 + lh * 8;
  v8us lo = *(const v8us*)(q);       // K: k0+lh*8    .. +7
  v8us hi = *(const v8us*)(q + 16);  // K: k0+16+lh*8 .. +7
  v16us r = __builtin_shufflevector(lo, hi,
      0, 1, 2, 3, 4, 5, 6, 7, 8, 9, 10, 11, 12, 13, 14, 15);
  return __builtin_bit_cast(v16bf, r);
}

// B-matrix 32x16 bf16 layout: lane holds column N = lane&15; lane-half lh:
// element e (0..15): K = k0 + (e>=8 ? 16 : 0) + lh*8 + (e&7) (same as A).
// Weights stored row-major W[col][k] (B = W^T) -> two b128 loads.
__device__ __forceinline__ v16bf load_b_frag(const unsigned short* __restrict__ colp,
                                             int k0, int lh) {
  const unsigned short* q = colp + k0 + (lh << 3);
  v8us lo = *(const v8us*)(q);
  v8us hi = *(const v8us*)(q + 16);
  v16us r = __builtin_shufflevector(lo, hi,
      0, 1, 2, 3, 4, 5, 6, 7, 8, 9, 10, 11, 12, 13, 14, 15);
  return __builtin_bit_cast(v16bf, r);
}

// ---- Fused dual GEMM: out[N][4E] = A0 @ W0^T + A1 @ W1^T + b0 + b1 ---------
// All matrix operands bf16. Block = 256 threads = 8 wave32s. Each wave owns a
// 32(M)x16(N) output: two f32 accumulators share each B fragment, so every
// B load feeds two v_wmma_f32_16x16x32_bf16. K=1024 in 32-deep steps.
__global__ void __launch_bounds__(256)
gemm_dual_bf16(const unsigned short* __restrict__ A0,
               const unsigned short* __restrict__ A1,
               const unsigned short* __restrict__ W0,
               const unsigned short* __restrict__ W1,
               const float* __restrict__ b0, const float* __restrict__ b1,
               float* __restrict__ out) {
  const int lane = threadIdx.x & 31;
  const int wave = threadIdx.x >> 5;
  const int lh   = lane >> 4;      // lane half (0/1)
  const int ll   = lane & 15;
  const int m0   = blockIdx.y << 5;                 // 32-row tile
  const int n0   = (blockIdx.x << 7) + (wave << 4); // 16-col tile per wave

  const unsigned short* a0lo = A0 + (size_t)(m0 + ll) * E_DIM;
  const unsigned short* a0hi = a0lo + (size_t)16 * E_DIM;
  const unsigned short* a1lo = A1 + (size_t)(m0 + ll) * E_DIM;
  const unsigned short* a1hi = a1lo + (size_t)16 * E_DIM;
  const unsigned short* w0p  = W0 + (size_t)(n0 + ll) * E_DIM;
  const unsigned short* w1p  = W1 + (size_t)(n0 + ll) * E_DIM;

  v8f acc0 = {}, acc1 = {};
#pragma unroll 2
  for (int k0 = 0; k0 < E_DIM; k0 += 32) {
    v16bf bf0 = load_b_frag(w0p, k0, lh);
    v16bf a00 = load_a_frag(a0lo, k0, lh);
    v16bf a01 = load_a_frag(a0hi, k0, lh);
    acc0 = __builtin_amdgcn_wmma_f32_16x16x32_bf16(
        false, a00, false, bf0, (short)0, acc0, false, false);
    acc1 = __builtin_amdgcn_wmma_f32_16x16x32_bf16(
        false, a01, false, bf0, (short)0, acc1, false, false);
    v16bf bf1 = load_b_frag(w1p, k0, lh);
    v16bf a10 = load_a_frag(a1lo, k0, lh);
    v16bf a11 = load_a_frag(a1hi, k0, lh);
    acc0 = __builtin_amdgcn_wmma_f32_16x16x32_bf16(
        false, a10, false, bf1, (short)0, acc0, false, false);
    acc1 = __builtin_amdgcn_wmma_f32_16x16x32_bf16(
        false, a11, false, bf1, (short)0, acc1, false, false);
  }

  // D layout: VGPR r, lanes 0-15 -> M=r, lanes 16-31 -> M=8+r; N = lane&15
  const int   col  = n0 + ll;
  const float bias = b0[col] + b1[col];
  float* op0 = out + (size_t)(m0 + lh * 8) * G_DIM + col;
  float* op1 = op0 + (size_t)16 * G_DIM;
#pragma unroll
  for (int r = 0; r < 8; ++r) {
    op0[(size_t)r * G_DIM] = acc0[r] + bias;
    op1[(size_t)r * G_DIM] = acc1[r] + bias;
  }
}

// ---- Elementwise lower LSTM cell (c_l already pre-masked by prev step) -----
// Writes h_l in fp32 (for stop-dot / masking) and bf16 (upper-GEMM operand).
__global__ void __launch_bounds__(256)
cell_lower(const float* __restrict__ g, float* __restrict__ c_l,
           float* __restrict__ h_l, unsigned short* __restrict__ h_l_bf) {
  const int idx = blockIdx.x * 256 + threadIdx.x;  // 0 .. N*E-1
  const int n = idx >> 10, e = idx & (E_DIM - 1);
  const float* gr = g + (size_t)n * G_DIM;
  const float zi = gr[e], zf = gr[E_DIM + e], zg = gr[2 * E_DIM + e],
              zo = gr[3 * E_DIM + e];
  const float c  = c_l[idx];
  const float cn = sigm(zf) * c + sigm(zi) * tanhf(zg);
  const float h  = sigm(zo) * tanhf(cn);
  c_l[idx]    = cn;
  h_l[idx]    = h;
  h_l_bf[idx] = f2bf(h);
}

// ---- stop[n] = sigmoid(h_l[n] . W_stop + b_stop), one WG per batch row -----
__global__ void __launch_bounds__(256)
stop_kernel(const float* __restrict__ h_l, const float* __restrict__ Ws,
            const float* __restrict__ bs, float* __restrict__ stop) {
  __shared__ float red[256];
  const int n = blockIdx.x, t = threadIdx.x;
  const float* hr = h_l + (size_t)n * E_DIM;
  float s = 0.0f;
#pragma unroll
  for (int j = 0; j < 4; ++j) {
    const int e = t + j * 256;
    s += hr[e] * Ws[e];
  }
  red[t] = s;
  __syncthreads();
  for (int w = 128; w > 0; w >>= 1) {
    if (t < w) red[t] += red[t + w];
    __syncthreads();
  }
  if (t == 0) stop[n] = sigm(red[0] + bs[0]);
}

// ---- Upper cell + stop-blend; also pre-mask lower state for next step ------
// Emits bf16 operands for the next step's GEMMs (hl_m_bf, h_u_bf).
__global__ void __launch_bounds__(256)
cell_upper(const float* __restrict__ g, const float* __restrict__ stop,
           const float* __restrict__ h_l, float* __restrict__ c_l,
           unsigned short* __restrict__ hl_m_bf, float* __restrict__ h_u,
           unsigned short* __restrict__ h_u_bf, float* __restrict__ c_u) {
  const int idx = blockIdx.x * 256 + threadIdx.x;
  const int n = idx >> 10, e = idx & (E_DIM - 1);
  const float* gr = g + (size_t)n * G_DIM;
  const float zi = gr[e], zf = gr[E_DIM + e], zg = gr[2 * E_DIM + e],
              zo = gr[3 * E_DIM + e];
  const float c  = c_u[idx];
  const float cn = sigm(zf) * c + sigm(zi) * tanhf(zg);
  const float hn = sigm(zo) * tanhf(cn);
  const float s = stop[n], r = 1.0f - s;
  const float hu = s * hn + r * h_u[idx];
  h_u[idx]    = hu;
  h_u_bf[idx] = f2bf(hu);
  c_u[idx]    = s * cn + r * c;
  // pre-mask lower state with (1 - stop) for next step's GEMM / cell
  hl_m_bf[idx] = f2bf(h_l[idx] * r);
  c_l[idx] *= r;
}

// ---- fp32 -> bf16 bulk conversion (4 elems/thread, b128 in / b64 out) ------
__global__ void __launch_bounds__(256)
cvt_bf16(const float* __restrict__ src, unsigned short* __restrict__ dst) {
  const int idx = (blockIdx.x * 256 + threadIdx.x) * 4;
  float4 f = *(const float4*)(src + idx);
  v4us r;
  r[0] = f2bf(f.x); r[1] = f2bf(f.y); r[2] = f2bf(f.z); r[3] = f2bf(f.w);
  *(v4us*)(dst + idx) = r;
}

__global__ void __launch_bounds__(256)
init_state(unsigned short* __restrict__ hl_m_bf, float* __restrict__ c_l,
           float* __restrict__ h_u, unsigned short* __restrict__ h_u_bf,
           float* __restrict__ c_u) {
  const int idx = blockIdx.x * 256 + threadIdx.x;
  hl_m_bf[idx] = 0; c_l[idx] = 0.0f;
  h_u[idx] = 0.0f;  h_u_bf[idx] = 0; c_u[idx] = 0.0f;
}

extern "C" void kernel_launch(void* const* d_in, const int* in_sizes, int n_in,
                              void* d_out, int out_size, void* d_ws,
                              size_t ws_size, hipStream_t stream) {
  const float* x      = (const float*)d_in[0];   // [S,N,E]
  const float* Wih_l  = (const float*)d_in[1];   // [4E,E]
  const float* Whh_l  = (const float*)d_in[2];
  const float* b_ih_l = (const float*)d_in[3];
  const float* b_hh_l = (const float*)d_in[4];
  const float* Wih_u  = (const float*)d_in[5];
  const float* Whh_u  = (const float*)d_in[6];
  const float* b_ih_u = (const float*)d_in[7];
  const float* b_hh_u = (const float*)d_in[8];
  const float* W_stop = (const float*)d_in[9];   // [1,E]
  const float* b_stop = (const float*)d_in[10];  // [1]

  // Workspace carve-out
  char* ws = (char*)d_ws;
  size_t off = 0;
  auto carve = [&](size_t bytes) -> void* {
    void* p = ws + off;
    off += (bytes + 255) & ~(size_t)255;
    return p;
  };
  const size_t WBYTES  = (size_t)G_DIM * E_DIM * sizeof(unsigned short);
  const size_t SBYTES  = (size_t)N_B * E_DIM * sizeof(float);
  const size_t SBYTES2 = (size_t)N_B * E_DIM * sizeof(unsigned short);
  unsigned short* Wl0    = (unsigned short*)carve(WBYTES);  // bf16 W_ih_l
  unsigned short* Wl1    = (unsigned short*)carve(WBYTES);  // bf16 W_hh_l
  unsigned short* Wu0    = (unsigned short*)carve(WBYTES);  // bf16 W_ih_u
  unsigned short* Wu1    = (unsigned short*)carve(WBYTES);  // bf16 W_hh_u
  unsigned short* x_bf   = (unsigned short*)carve(SBYTES2); // bf16 x_t staging
  float*          h_l    = (float*)carve(SBYTES);
  unsigned short* h_l_bf = (unsigned short*)carve(SBYTES2);
  unsigned short* hl_m_bf= (unsigned short*)carve(SBYTES2); // (1-stop)*h_l
  float*          c_l    = (float*)carve(SBYTES);
  float*          h_u    = (float*)carve(SBYTES);
  unsigned short* h_u_bf = (unsigned short*)carve(SBYTES2);
  float*          c_u    = (float*)carve(SBYTES);
  float*          gates  = (float*)carve((size_t)N_B * G_DIM * sizeof(float));
  float*          stopb  = (float*)carve((size_t)N_B * sizeof(float));

  // One-time per call: weight conversion + state init (deterministic)
  const int wCvtBlocks = (G_DIM * E_DIM) / (256 * 4);  // 4096
  const int xCvtBlocks = (N_B * E_DIM) / (256 * 4);    // 256
  cvt_bf16<<<wCvtBlocks, 256, 0, stream>>>(Wih_l, Wl0);
  cvt_bf16<<<wCvtBlocks, 256, 0, stream>>>(Whh_l, Wl1);
  cvt_bf16<<<wCvtBlocks, 256, 0, stream>>>(Wih_u, Wu0);
  cvt_bf16<<<wCvtBlocks, 256, 0, stream>>>(Whh_u, Wu1);
  init_state<<<(N_B * E_DIM) / 256, 256, 0, stream>>>(hl_m_bf, c_l, h_u,
                                                      h_u_bf, c_u);

  const dim3 ggrid(G_DIM / 128, N_B / 32);  // 32 x 8 = 256 WGs, 8 waves each
  const int  eBlocks = (N_B * E_DIM) / 256; // 1024

  for (int t = 0; t < S_LEN; ++t) {
    const float* xt = x + (size_t)t * N_B * E_DIM;
    cvt_bf16<<<xCvtBlocks, 256, 0, stream>>>(xt, x_bf);
    // gates_l = x_t @ Wih_l^T + hl_masked @ Whh_l^T + b_ih_l + b_hh_l
    gemm_dual_bf16<<<ggrid, 256, 0, stream>>>(x_bf, hl_m_bf, Wl0, Wl1,
                                              b_ih_l, b_hh_l, gates);
    cell_lower<<<eBlocks, 256, 0, stream>>>(gates, c_l, h_l, h_l_bf);
    stop_kernel<<<N_B, 256, 0, stream>>>(h_l, W_stop, b_stop, stopb);
    // gates_u = h_l @ Wih_u^T + h_u @ Whh_u^T + b_ih_u + b_hh_u
    gemm_dual_bf16<<<ggrid, 256, 0, stream>>>(h_l_bf, h_u_bf, Wu0, Wu1,
                                              b_ih_u, b_hh_u, gates);
    cell_upper<<<eBlocks, 256, 0, stream>>>(gates, stopb, h_l, c_l, hl_m_bf,
                                            h_u, h_u_bf, c_u);
  }

  hipMemcpyAsync(d_out, h_u, SBYTES, hipMemcpyDeviceToDevice, stream);
}
